// MultilayerGRU_68126771249426
// MI455X (gfx1250) — compile-verified, matrix operand order checked by reading, not run on hardware
//
#include <hip/hip_runtime.h>
#include <stdint.h>

#define B_ 32
#define S_ 2048
#define I_ 128
#define H_ 256
#define O_ 128
#define GST ((size_t)S_ * B_ * H_) // per-gate stride in floats
#define HPAD 264                   // padded H row (bf16/f32 elems) to dodge LDS bank conflicts
#define GSTEP (B_ * H_)            // floats per gate per timestep (8192)

typedef __bf16 bf16_t;
typedef bf16_t bfrag __attribute__((ext_vector_type(16))); // 8 VGPRs of bf16 pairs
typedef float  ffrag __attribute__((ext_vector_type(8)));  // 8 VGPRs f32 accumulator
typedef int    v4i_t __attribute__((ext_vector_type(4)));

static __device__ __forceinline__ uint16_t f2bf(float f) {
  uint32_t u = __builtin_bit_cast(uint32_t, f);
  u += 0x7FFFu + ((u >> 16) & 1u); // round-to-nearest-even
  return (uint16_t)(u >> 16);
}

// Load one 16x32 bf16 A-fragment (or 32x16 B-fragment) for lane (ln, half)
// from a row-major [16][K] slice: two b128 loads per lane, per ISA 7.12.2 layout.
static __device__ __forceinline__ bfrag load_frag(const uint16_t* __restrict__ rowp,
                                                  int kf, int half) {
  const uint4* p = (const uint4*)rowp; // row is 16B aligned
  union { bfrag v; uint4 q[2]; } f;
  f.q[0] = p[kf * 4 + half];     // K = kf*32 + half*8 .. +7
  f.q[1] = p[kf * 4 + half + 2]; // K = kf*32 + 16 + half*8 .. +7
  return f.v;
}

static __device__ __forceinline__ ffrag wmma_bf16(bfrag a, bfrag b, ffrag c) {
  return __builtin_amdgcn_wmma_f32_16x16x32_bf16(false, a, false, b, (short)0, c,
                                                 false, false);
}

static __device__ __forceinline__ float sigm(float x) {
  return 1.f / (1.f + __expf(-x));
}

// ---- gfx1250 async global->LDS copy (ASYNCcnt path), with safe fallback ----
#if defined(__has_builtin)
#if __has_builtin(__builtin_amdgcn_global_load_async_to_lds_b128)
#define HAS_ASYNC_LDS 1
#endif
#endif
#ifndef HAS_ASYNC_LDS
#define HAS_ASYNC_LDS 0
#endif

static __device__ __forceinline__ void cp16_async(const float* g, float* l) {
#if HAS_ASYNC_LDS
  __builtin_amdgcn_global_load_async_to_lds_b128(
      (v4i_t __attribute__((address_space(1)))*)(g),
      (v4i_t __attribute__((address_space(3)))*)(l), 0, 0);
#else
  *(uint4*)l = *(const uint4*)g;
#endif
}
static __device__ __forceinline__ void async_drain() {
#if HAS_ASYNC_LDS
  asm volatile("s_wait_asynccnt 0" ::: "memory");
#endif
}

// ---------------------------------------------------------------- converts
__global__ void k_f32_to_bf16(const float* __restrict__ src,
                              uint16_t* __restrict__ dst, int n) {
  int i = blockIdx.x * blockDim.x + threadIdx.x;
  if (i < n) dst[i] = f2bf(src[i]);
}

// -------------------------------------------- batched input projections
// C[m][n] = sum_k A[m][k] * W[n][k], A:[65536][K] bf16 (m = b*2048+s),
// W:[256][K] bf16. Output scattered time-major: gates[gate][s][b][n] (f32).
__global__ void __launch_bounds__(256) k_xproj(const uint16_t* __restrict__ A, int K,
                                               const uint16_t* __restrict__ Wz,
                                               const uint16_t* __restrict__ Wr,
                                               const uint16_t* __restrict__ Wg,
                                               float* __restrict__ gates) {
  const int tid = threadIdx.x;
  const int lane = tid & 31, wid = tid >> 5;
  const int ln = lane & 15, half = lane >> 4;
  const int wm = wid & 3, wn = wid >> 2;          // 8 waves: 4 (M) x 2 (N)
  const int mb = blockIdx.x * 128 + wm * 32;      // 32-row macro tile per wave
  const int nb = blockIdx.y * 64 + wn * 32;       // 32-col macro tile per wave
  const int gate = blockIdx.z;
  const uint16_t* W = (gate == 0) ? Wz : ((gate == 1) ? Wr : Wg);

  ffrag acc[2][2] = {};
  const int nk = K >> 5;
  for (int kf = 0; kf < nk; ++kf) {
    bfrag a0 = load_frag(A + (size_t)(mb + ln) * K, kf, half);
    bfrag a1 = load_frag(A + (size_t)(mb + 16 + ln) * K, kf, half);
    bfrag b0 = load_frag(W + (size_t)(nb + ln) * K, kf, half);
    bfrag b1 = load_frag(W + (size_t)(nb + 16 + ln) * K, kf, half);
    acc[0][0] = wmma_bf16(a0, b0, acc[0][0]);
    acc[0][1] = wmma_bf16(a0, b1, acc[0][1]);
    acc[1][0] = wmma_bf16(a1, b0, acc[1][0]);
    acc[1][1] = wmma_bf16(a1, b1, acc[1][1]);
  }
  float* gout = gates + (size_t)gate * GST;
#pragma unroll
  for (int tm = 0; tm < 2; ++tm)
#pragma unroll
    for (int tn = 0; tn < 2; ++tn)
#pragma unroll
      for (int r = 0; r < 8; ++r) {
        int m = mb + tm * 16 + half * 8 + r; // C layout: lanes16-31 hold M=8..15
        int n = nb + tn * 16 + ln;
        int b = m >> 11, s = m & 2047;       // m = b*S + s
        gout[((size_t)s * B_ + b) * H_ + n] = acc[tm][tn][r];
      }
}

// -------------------------------------------------- final output projection
__global__ void __launch_bounds__(256) k_outproj(const uint16_t* __restrict__ A,
                                                 const uint16_t* __restrict__ Wo,
                                                 const float* __restrict__ bout,
                                                 float* __restrict__ out) {
  const int tid = threadIdx.x;
  const int lane = tid & 31, wid = tid >> 5;
  const int ln = lane & 15, half = lane >> 4;
  const int wm = wid & 3, wn = wid >> 2;
  const int mb = blockIdx.x * 128 + wm * 32;
  const int nb = blockIdx.y * 64 + wn * 32;

  ffrag acc[2][2] = {};
  for (int kf = 0; kf < (H_ >> 5); ++kf) {
    bfrag a0 = load_frag(A + (size_t)(mb + ln) * H_, kf, half);
    bfrag a1 = load_frag(A + (size_t)(mb + 16 + ln) * H_, kf, half);
    bfrag b0 = load_frag(Wo + (size_t)(nb + ln) * H_, kf, half);
    bfrag b1 = load_frag(Wo + (size_t)(nb + 16 + ln) * H_, kf, half);
    acc[0][0] = wmma_bf16(a0, b0, acc[0][0]);
    acc[0][1] = wmma_bf16(a0, b1, acc[0][1]);
    acc[1][0] = wmma_bf16(a1, b0, acc[1][0]);
    acc[1][1] = wmma_bf16(a1, b1, acc[1][1]);
  }
#pragma unroll
  for (int tm = 0; tm < 2; ++tm)
#pragma unroll
    for (int tn = 0; tn < 2; ++tn)
#pragma unroll
      for (int r = 0; r < 8; ++r) {
        int m = mb + tm * 16 + half * 8 + r;
        int n = nb + tn * 16 + ln;
        out[(size_t)m * O_ + n] = acc[tm][tn][r] + bout[n];
      }
}

// ------------------------------------------------------ persistent GRU scan
// One workgroup (16 waves). Hidden state (32x256) lives in LDS as f32 master +
// bf16 WMMA copy. Each wave owns one 16-wide N slice; its recurrent-weight
// B-fragments (3 gates x 8 k-frags) are register resident across all 2048 steps.
// Gate pre-activations for step t+1 are streamed L2->LDS with async b128 copies
// (double buffered, drained with s_wait_asynccnt) while step t's WMMAs run.
__global__ void __launch_bounds__(512) k_scan(
    const float* __restrict__ gates,
    const uint16_t* __restrict__ Wzh, const uint16_t* __restrict__ Wrh,
    const uint16_t* __restrict__ Wgh,
    const float* __restrict__ bz, const float* __restrict__ br,
    const float* __restrict__ bg,
    const float* __restrict__ h0, int layer,
    uint16_t* __restrict__ seqout, float* __restrict__ hfin) {
  __shared__ __align__(16) float    hS[B_ * HPAD];
  __shared__ __align__(16) uint16_t hB[B_ * HPAD];
  __shared__ __align__(16) uint16_t rhB[B_ * HPAD];
  __shared__ __align__(16) float    gL[2][3 * GSTEP]; // double-buffered gate tiles

  const int tid = threadIdx.x;
  const int lane = tid & 31, ni = tid >> 5; // wave id == N tile index (0..15)
  const int ln = lane & 15, half = lane >> 4;
  const int n = ni * 16 + ln;

  bfrag wz[8], wr[8], wg[8];
#pragma unroll
  for (int kf = 0; kf < 8; ++kf) {
    wz[kf] = load_frag(Wzh + (size_t)n * H_, kf, half);
    wr[kf] = load_frag(Wrh + (size_t)n * H_, kf, half);
    wg[kf] = load_frag(Wgh + (size_t)n * H_, kf, half);
  }
  const float bzv = bz[n], brv = br[n], bgv = bg[n];

  // init hidden state
  for (int i = tid; i < B_ * H_; i += 512) {
    int b = i >> 8, hh = i & 255;
    float v = h0[(size_t)b * (2 * H_) + (size_t)layer * H_ + hh];
    hS[b * HPAD + hh] = v;
    hB[b * HPAD + hh] = f2bf(v);
  }
  // prefetch gate tile for t=0 (3 x 8192 floats = 6144 b128 per step)
#pragma unroll
  for (int j = 0; j < 12; ++j) {
    int i = tid + j * 512;            // uint4 index within step tile
    int g = i >> 11, off = (i & 2047) << 2;
    cp16_async(gates + (size_t)g * GST + off, &gL[0][g * GSTEP + off]);
  }
  async_drain();
  __syncthreads();

  for (int t = 0; t < S_; ++t) {
    const float* gcur = gL[t & 1];
    // kick off async copy of step t+1 into the other buffer
    if (t + 1 < S_) {
      float* gnxt = gL[(t + 1) & 1];
      const float* gsrc = gates + (size_t)(t + 1) * GSTEP;
#pragma unroll
      for (int j = 0; j < 12; ++j) {
        int i = tid + j * 512;
        int g = i >> 11, off = (i & 2047) << 2;
        cp16_async(gsrc + (size_t)g * GST + off, &gnxt[g * GSTEP + off]);
      }
    }

    // ---- phase 1: Z and R gates -------------------------------------
    ffrag az[2], ar[2];
#pragma unroll
    for (int mi = 0; mi < 2; ++mi)
#pragma unroll
      for (int r = 0; r < 8; ++r) {
        int b = mi * 16 + half * 8 + r;
        az[mi][r] = gcur[0 * GSTEP + b * H_ + n];
        ar[mi][r] = gcur[1 * GSTEP + b * H_ + n];
      }
#pragma unroll
    for (int kf = 0; kf < 8; ++kf) {
#pragma unroll
      for (int mi = 0; mi < 2; ++mi) {
        bfrag ha = load_frag((const uint16_t*)&hB[(mi * 16 + ln) * HPAD], kf, half);
        az[mi] = wmma_bf16(ha, wz[kf], az[mi]);
        ar[mi] = wmma_bf16(ha, wr[kf], ar[mi]);
      }
    }
    float Z[2][8];
#pragma unroll
    for (int mi = 0; mi < 2; ++mi)
#pragma unroll
      for (int r = 0; r < 8; ++r) {
        int b = mi * 16 + half * 8 + r;
        Z[mi][r] = sigm(az[mi][r] + bzv);
        float R = sigm(ar[mi][r] + brv);
        rhB[b * HPAD + n] = f2bf(R * hS[b * HPAD + n]);
      }
    __syncthreads(); // rh visible; phase-1 hB reads retired

    // ---- phase 2: G gate and state update ---------------------------
    ffrag ag[2];
#pragma unroll
    for (int mi = 0; mi < 2; ++mi)
#pragma unroll
      for (int r = 0; r < 8; ++r) {
        int b = mi * 16 + half * 8 + r;
        ag[mi][r] = gcur[2 * GSTEP + b * H_ + n];
      }
#pragma unroll
    for (int kf = 0; kf < 8; ++kf) {
#pragma unroll
      for (int mi = 0; mi < 2; ++mi) {
        bfrag ra = load_frag((const uint16_t*)&rhB[(mi * 16 + ln) * HPAD], kf, half);
        ag[mi] = wmma_bf16(ra, wg[kf], ag[mi]);
      }
    }
#pragma unroll
    for (int mi = 0; mi < 2; ++mi)
#pragma unroll
      for (int r = 0; r < 8; ++r) {
        int b = mi * 16 + half * 8 + r;
        float g = tanhf(ag[mi][r] + bgv);
        float hv = hS[b * HPAD + n];
        float z = Z[mi][r];
        float hn = z * hv + (1.f - z) * g;
        hS[b * HPAD + n] = hn;
        uint16_t hb = f2bf(hn);
        hB[b * HPAD + n] = hb;
        seqout[((size_t)b * S_ + t) * H_ + n] = hb; // next layer's bf16 A matrix
      }
    async_drain();   // t+1 gate tile resident
    __syncthreads(); // h_new + next gate tile visible before next step
  }
#pragma unroll
  for (int mi = 0; mi < 2; ++mi)
#pragma unroll
    for (int r = 0; r < 8; ++r) {
      int b = mi * 16 + half * 8 + r;
      hfin[(size_t)b * (2 * H_) + (size_t)layer * H_ + n] = hS[b * HPAD + n];
    }
}

// ------------------------------------------------------------------ driver
extern "C" void kernel_launch(void* const* d_in, const int* in_sizes, int n_in,
                              void* d_out, int out_size, void* d_ws, size_t ws_size,
                              hipStream_t stream) {
  (void)in_sizes; (void)n_in; (void)out_size; (void)ws_size;
  // JAX pytree (sorted-key) flattening order:
  const float* h0   = (const float*)d_in[0];
  const float* Wout = (const float*)d_in[1];
  const float* bout = (const float*)d_in[2];
  const float* Wgh0 = (const float*)d_in[3];
  const float* Wgx0 = (const float*)d_in[4];
  const float* Wrh0 = (const float*)d_in[5];
  const float* Wrx0 = (const float*)d_in[6];
  const float* Wzh0 = (const float*)d_in[7];
  const float* Wzx0 = (const float*)d_in[8];
  const float* bgh0 = (const float*)d_in[9];
  const float* brh0 = (const float*)d_in[10];
  const float* bzh0 = (const float*)d_in[11];
  const float* Wgh1 = (const float*)d_in[12];
  const float* Wgx1 = (const float*)d_in[13];
  const float* Wrh1 = (const float*)d_in[14];
  const float* Wrx1 = (const float*)d_in[15];
  const float* Wzh1 = (const float*)d_in[16];
  const float* Wzx1 = (const float*)d_in[17];
  const float* bgh1 = (const float*)d_in[18];
  const float* brh1 = (const float*)d_in[19];
  const float* bzh1 = (const float*)d_in[20];
  const float* x    = (const float*)d_in[21];

  char* ws = (char*)d_ws;
  uint16_t* xbf   = (uint16_t*)(ws);                 // 16 MiB: x as bf16
  uint16_t* wbf   = (uint16_t*)(ws + 16777216);      // 1.4 MiB: all weights bf16
  float*    gates = (float*)(ws + 18219008);         // 192 MiB: [3][S][B][H] f32
  uint16_t* seq0  = (uint16_t*)(ws + 219545600);     // 32 MiB: layer0 seq bf16
  uint16_t* seq1  = (uint16_t*)(ws + 253100032);     // 32 MiB: layer1 seq bf16

  uint16_t* wzx0b = wbf + 0;
  uint16_t* wrx0b = wbf + 32768;
  uint16_t* wgx0b = wbf + 65536;
  uint16_t* wzh0b = wbf + 98304;
  uint16_t* wrh0b = wbf + 163840;
  uint16_t* wgh0b = wbf + 229376;
  uint16_t* wzx1b = wbf + 294912;
  uint16_t* wrx1b = wbf + 360448;
  uint16_t* wgx1b = wbf + 425984;
  uint16_t* wzh1b = wbf + 491520;
  uint16_t* wrh1b = wbf + 557056;
  uint16_t* wgh1b = wbf + 622592;
  uint16_t* woutb = wbf + 688128;

  auto cv = [&](const float* s, uint16_t* dp, int n) {
    k_f32_to_bf16<<<(n + 255) / 256, 256, 0, stream>>>(s, dp, n);
  };
  cv(x, xbf, B_ * S_ * I_);
  cv(Wzx0, wzx0b, H_ * I_); cv(Wrx0, wrx0b, H_ * I_); cv(Wgx0, wgx0b, H_ * I_);
  cv(Wzh0, wzh0b, H_ * H_); cv(Wrh0, wrh0b, H_ * H_); cv(Wgh0, wgh0b, H_ * H_);
  cv(Wzx1, wzx1b, H_ * H_); cv(Wrx1, wrx1b, H_ * H_); cv(Wgx1, wgx1b, H_ * H_);
  cv(Wzh1, wzh1b, H_ * H_); cv(Wrh1, wrh1b, H_ * H_); cv(Wgh1, wgh1b, H_ * H_);
  cv(Wout, woutb, O_ * H_);

  float* outp = (float*)d_out;
  float* hfin = outp + (size_t)B_ * S_ * O_; // hidden_state (B,L,H) after out

  // Layer 0: batched projections then persistent scan
  k_xproj<<<dim3(512, 4, 3), 256, 0, stream>>>(xbf, I_, wzx0b, wrx0b, wgx0b, gates);
  k_scan<<<1, 512, 0, stream>>>(gates, wzh0b, wrh0b, wgh0b, bzh0, brh0, bgh0,
                                h0, 0, seq0, hfin);
  // Layer 1
  k_xproj<<<dim3(512, 4, 3), 256, 0, stream>>>(seq0, H_, wzx1b, wrx1b, wgx1b, gates);
  k_scan<<<1, 512, 0, stream>>>(gates, wzh1b, wrh1b, wgh1b, bzh1, brh1, bgh1,
                                h0, 1, seq1, hfin);
  // Output projection
  k_outproj<<<dim3(512, 2), 256, 0, stream>>>(seq1, woutb, bout, outp);
}
